// CrossModalAttention_8942121911104
// MI455X (gfx1250) — matrix operand order repaired and checked by reading, not compiled, not run on hardware
//
#include <hip/hip_runtime.h>
#include <hip/hip_bf16.h>
#include <math.h>

typedef _Float16 half_t;
typedef __attribute__((ext_vector_type(16))) _Float16 v16h;
typedef __attribute__((ext_vector_type(8)))  _Float16 v8h;
typedef __attribute__((ext_vector_type(8)))  float    v8f;
typedef int v4i_t __attribute__((vector_size(16)));   // matches builtin proto

// --------------------------------------------------------------------------
// CDNA5 async global->LDS copy (ASYNCcnt path), with safe fallback.
// Probe-learned prototype: (as1 int4*, as3 int4*, imm offset, imm cpol)
// --------------------------------------------------------------------------
#if defined(__has_builtin)
#if __has_builtin(__builtin_amdgcn_global_load_async_to_lds_b128)
#define HAVE_ASYNC_LDS 1
#endif
#endif
#ifndef HAVE_ASYNC_LDS
#define HAVE_ASYNC_LDS 0
#endif

__device__ __forceinline__ void cp32B_g2l(const half_t* __restrict__ g,
                                          half_t* __restrict__ l)
{
#if HAVE_ASYNC_LDS
    // two async b128 transfers (16B per lane each)
    __builtin_amdgcn_global_load_async_to_lds_b128(
        (__attribute__((address_space(1))) v4i_t*)g,
        (__attribute__((address_space(3))) v4i_t*)l, 0, 0);
    __builtin_amdgcn_global_load_async_to_lds_b128(
        (__attribute__((address_space(1))) v4i_t*)(g + 8),
        (__attribute__((address_space(3))) v4i_t*)(l + 8), 0, 0);
#else
    *(v16h*)l = *(const v16h*)g;
#endif
}

__device__ __forceinline__ void stage_fence()
{
#if HAVE_ASYNC_LDS
#if defined(__has_builtin) && __has_builtin(__builtin_amdgcn_s_wait_asynccnt)
    __builtin_amdgcn_s_wait_asynccnt(0);
#else
    asm volatile("s_wait_asynccnt 0x0" ::: "memory");
#endif
#endif
}

// ---------------------------------------------------------------------------
// LayerNorm (fp32 in) -> f16 out, one block per row
// ---------------------------------------------------------------------------
__global__ __launch_bounds__(256)
void ln_cast_kernel(const float* __restrict__ x, const float* __restrict__ g,
                    const float* __restrict__ b, half_t* __restrict__ y, int D)
{
    __shared__ float red[256];
    const size_t row = blockIdx.x;
    const float* xr = x + row * (size_t)D;
    half_t* yr = y + row * (size_t)D;
    const int tid = threadIdx.x;

    float s = 0.f, ss = 0.f;
    for (int i = tid; i < D; i += 256) { float v = xr[i]; s += v; ss += v * v; }

    red[tid] = s; __syncthreads();
    for (int o = 128; o > 0; o >>= 1) { if (tid < o) red[tid] += red[tid + o]; __syncthreads(); }
    const float mean = red[0] / (float)D;
    __syncthreads();
    red[tid] = ss; __syncthreads();
    for (int o = 128; o > 0; o >>= 1) { if (tid < o) red[tid] += red[tid + o]; __syncthreads(); }
    const float var = red[0] / (float)D - mean * mean;
    const float rstd = rsqrtf(var + 1e-5f);

    for (int i = tid; i < D; i += 256) {
        float v = (xr[i] - mean) * rstd * g[i] + b[i];
        yr[i] = (half_t)v;
    }
}

// ---------------------------------------------------------------------------
// Transpose+cast: W[R,C] fp32 -> Wt[C,R] f16   (R,C multiples of 32)
// ---------------------------------------------------------------------------
__global__ __launch_bounds__(256)
void tcast_kernel(const float* __restrict__ W, half_t* __restrict__ Wt, int R, int C)
{
    __shared__ float tile[32][33];
    const int c0 = blockIdx.x * 32;
    const int r0 = blockIdx.y * 32;
    const int tx = threadIdx.x, ty = threadIdx.y;
    for (int dy = 0; dy < 32; dy += 8)
        tile[ty + dy][tx] = W[(size_t)(r0 + ty + dy) * C + (c0 + tx)];
    __syncthreads();
    for (int dy = 0; dy < 32; dy += 8)
        Wt[(size_t)(c0 + ty + dy) * R + (r0 + tx)] = (half_t)tile[tx][ty + dy];
}

// Batched f16 transpose: X[z][R,C] -> Xt[z][C,R]
__global__ __launch_bounds__(256)
void transpose_f16_kernel(const half_t* __restrict__ X, half_t* __restrict__ Xt, int R, int C)
{
    __shared__ half_t tile[32][34];
    const size_t zoff = (size_t)blockIdx.z * (size_t)R * (size_t)C;
    const int c0 = blockIdx.x * 32;
    const int r0 = blockIdx.y * 32;
    const int tx = threadIdx.x, ty = threadIdx.y;
    for (int dy = 0; dy < 32; dy += 8)
        tile[ty + dy][tx] = X[zoff + (size_t)(r0 + ty + dy) * C + (c0 + tx)];
    __syncthreads();
    for (int dy = 0; dy < 32; dy += 8)
        Xt[zoff + (size_t)(c0 + ty + dy) * R + (r0 + tx)] = tile[tx][ty + dy];
}

// ---------------------------------------------------------------------------
// Row softmax, in place on f16, L == 2048 (8 elems/thread), one block per row
// ---------------------------------------------------------------------------
__global__ __launch_bounds__(256)
void softmax_rows_kernel(half_t* __restrict__ S, int L)
{
    __shared__ float red[256];
    half_t* p = S + (size_t)blockIdx.x * (size_t)L;
    const int tid = threadIdx.x;

    float v[8];
    float mx = -3.402823466e38f;
#pragma unroll
    for (int i = 0; i < 8; ++i) { v[i] = (float)p[tid + i * 256]; mx = fmaxf(mx, v[i]); }
    red[tid] = mx; __syncthreads();
    for (int o = 128; o > 0; o >>= 1) { if (tid < o) red[tid] = fmaxf(red[tid], red[tid + o]); __syncthreads(); }
    mx = red[0]; __syncthreads();

    float s = 0.f;
#pragma unroll
    for (int i = 0; i < 8; ++i) { v[i] = __expf(v[i] - mx); s += v[i]; }
    red[tid] = s; __syncthreads();
    for (int o = 128; o > 0; o >>= 1) { if (tid < o) red[tid] += red[tid + o]; __syncthreads(); }
    const float inv = 1.0f / red[0];
#pragma unroll
    for (int i = 0; i < 8; ++i) p[tid + i * 256] = (half_t)(v[i] * inv);
}

// ---------------------------------------------------------------------------
// WMMA GEMM:  C[M,N] = alpha * (A[M,K] @ Bt[N,K]^T + bias[N])
// A,Bt f16 row-major (K contiguous). Block tile 128x128, BK=32, 8 waves.
// Double-buffered LDS staging via async global->LDS copies (ASYNCcnt),
// each wave: 64x32 tile = 4x2 v_wmma_f32_16x16x32_f16 accumulators.
// M,N multiples of 128; K multiple of 32. Batched via blockIdx.z strides.
// ---------------------------------------------------------------------------
template<typename OutT, bool HasBias>
__global__ __launch_bounds__(256)
void gemm_wmma_kernel(const half_t* __restrict__ A, const half_t* __restrict__ Bt,
                      const float* __restrict__ bias, OutT* __restrict__ C,
                      int M, int N, int K,
                      long long sA, long long sB, long long sC, float alpha)
{
    constexpr int BM = 128, BN = 128, BK = 32, LDS_S = BK + 8; // 40 halfs/row
    __shared__ alignas(32) half_t As[2][BM * LDS_S];
    __shared__ alignas(32) half_t Bs[2][BN * LDS_S];

    const long long bz = blockIdx.z;
    A  += bz * sA;
    Bt += bz * sB;
    C  += bz * sC;

    const int m0   = blockIdx.y * BM;
    const int n0   = blockIdx.x * BN;
    const int tid  = threadIdx.x;
    const int lane = tid & 31;
    const int wave = tid >> 5;
    const int wm   = (wave >> 2) * 64;   // wave M offset (2 waves in M)
    const int wn   = (wave & 3) * 32;    // wave N offset (4 waves in N)

    const int lr  = lane & 15;           // A row-in-tile / B col-in-tile
    const int akb = (lane >> 4) * 16;    // A k base per half-wave
    const int bkb = (lane >> 4) * 16;    // B k base per half-wave

    // staging indices: each thread moves 16 halfs of A and 16 of B per K step
    const int grow = tid >> 1;           // 0..127
    const int gcol = (tid & 1) * 16;     // 0 or 16

    const half_t* Ag = A  + (size_t)(m0 + grow) * K + gcol;  // + k0
    const half_t* Bg = Bt + (size_t)(n0 + grow) * K + gcol;  // + k0
    half_t* Al0 = &As[0][grow * LDS_S + gcol];
    half_t* Bl0 = &Bs[0][grow * LDS_S + gcol];
    constexpr int BUF = BM * LDS_S;      // halfs per buffer

    v8f acc[4][2] = {};

    const int nsteps = K / BK;
    // prologue: stage step 0 into buffer 0
    cp32B_g2l(Ag, Al0);
    cp32B_g2l(Bg, Bl0);

    for (int s = 0; s < nsteps; ++s) {
        const int cur = s & 1;
        stage_fence();        // own async copies done (LDS data landed)
        __syncthreads();      // everyone's copies visible; prev reads done

        if (s + 1 < nsteps) { // stage next tile into the other buffer
            const int kk = (s + 1) * BK;
            cp32B_g2l(Ag + kk, Al0 + (cur ^ 1) * BUF);
            cp32B_g2l(Bg + kk, Bl0 + (cur ^ 1) * BUF);
        }

        const half_t* Ab = &As[cur][0];
        const half_t* Bb = &Bs[cur][0];
        v16h af[4], bf[2];
#pragma unroll
        for (int i = 0; i < 4; ++i) {
            const half_t* p = Ab + (wm + i * 16 + lr) * LDS_S + akb;
            v8h lo = *(const v8h*)p;          // K = akb .. akb+7
            v8h hi = *(const v8h*)(p + 8);    // K = akb+8 .. akb+15
            af[i] = __builtin_shufflevector(lo, hi,
                        0,1,2,3,4,5,6,7,8,9,10,11,12,13,14,15);
        }
#pragma unroll
        for (int j = 0; j < 2; ++j) {
            bf[j] = *(const v16h*)(Bb + (wn + j * 16 + lr) * LDS_S + bkb);
        }
#pragma unroll
        for (int i = 0; i < 4; ++i)
#pragma unroll
            for (int j = 0; j < 2; ++j)
                acc[i][j] = __builtin_amdgcn_wmma_f32_16x16x32_f16(
                    false, af[i], false, bf[j], (short)0, acc[i][j], false, false);

        __syncthreads();      // all reads of `cur` done before it is restaged
    }

    // Epilogue: C layout — lane 0-15: N=lane, M=vgpr; lane 16-31: M=8+vgpr
    const int rbase = (lane >> 4) * 8;
#pragma unroll
    for (int i = 0; i < 4; ++i) {
#pragma unroll
        for (int j = 0; j < 2; ++j) {
            const int col = n0 + wn + j * 16 + lr;
            const float bval = HasBias ? bias[col] : 0.0f;
#pragma unroll
            for (int v = 0; v < 8; ++v) {
                const int row = m0 + wm + i * 16 + rbase + v;
                const float r = (acc[i][j][v] + bval) * alpha;
                C[(size_t)row * N + col] = (OutT)r;
            }
        }
    }
}

// ---------------------------------------------------------------------------
extern "C" void kernel_launch(void* const* d_in, const int* in_sizes, int n_in,
                              void* d_out, int out_size, void* d_ws, size_t ws_size,
                              hipStream_t stream) {
    const float* m1   = (const float*)d_in[0];
    const float* m2   = (const float*)d_in[1];
    const float* ln1g = (const float*)d_in[2];
    const float* ln1b = (const float*)d_in[3];
    const float* ln2g = (const float*)d_in[4];
    const float* ln2b = (const float*)d_in[5];
    const float* Wq   = (const float*)d_in[6];
    const float* bq   = (const float*)d_in[7];
    const float* Wk   = (const float*)d_in[8];
    const float* bk   = (const float*)d_in[9];
    const float* Wv   = (const float*)d_in[10];
    const float* bv   = (const float*)d_in[11];
    const float* Wo   = (const float*)d_in[12];
    const float* bo   = (const float*)d_in[13];
    float* out = (float*)d_out;

    constexpr int B = 8, L1 = 2048, L2 = 2048, D1 = 1024, D2 = 768, E = 1024;
    const int M1 = B * L1;   // 16384 query rows
    const int M2 = B * L2;   // 16384 key rows

    // workspace carving (256B aligned)
    char* w = (char*)d_ws;
    auto carve = [&](size_t bytes) -> char* {
        char* p = w; w += (bytes + 255) & ~(size_t)255; return p;
    };
    half_t* X1n = (half_t*)carve((size_t)M1 * D1 * 2);
    half_t* X2n = (half_t*)carve((size_t)M2 * D2 * 2);
    half_t* Wqt = (half_t*)carve((size_t)E * D1 * 2);
    half_t* Wkt = (half_t*)carve((size_t)E * D2 * 2);
    half_t* Wvt = (half_t*)carve((size_t)E * D2 * 2);
    half_t* Wot = (half_t*)carve((size_t)D1 * E * 2);
    half_t* Qh  = (half_t*)carve((size_t)M1 * E * 2);
    half_t* Kh  = (half_t*)carve((size_t)M2 * E * 2);
    half_t* Vh  = (half_t*)carve((size_t)M2 * E * 2);
    half_t* Vt  = (half_t*)carve((size_t)B * E * L2 * 2);
    half_t* Sh  = (half_t*)carve((size_t)B * L1 * L2 * 2);
    half_t* Ctx = (half_t*)carve((size_t)M1 * E * 2);
    (void)ws_size; (void)in_sizes; (void)n_in; (void)out_size;

    const dim3 tb(32, 8);

    // 1) LayerNorm + f16 cast
    ln_cast_kernel<<<M1, 256, 0, stream>>>(m1, ln1g, ln1b, X1n, D1);
    ln_cast_kernel<<<M2, 256, 0, stream>>>(m2, ln2g, ln2b, X2n, D2);

    // 2) Weight transpose+cast: W[D,E] -> Wt[E,D] f16
    tcast_kernel<<<dim3(E / 32, D1 / 32), tb, 0, stream>>>(Wq, Wqt, D1, E);
    tcast_kernel<<<dim3(E / 32, D2 / 32), tb, 0, stream>>>(Wk, Wkt, D2, E);
    tcast_kernel<<<dim3(E / 32, D2 / 32), tb, 0, stream>>>(Wv, Wvt, D2, E);
    tcast_kernel<<<dim3(D1 / 32, E / 32), tb, 0, stream>>>(Wo, Wot, E, D1);

    // 3) Projections (softmax scale folded into Q)
    const float scale = 1.0f / sqrtf((float)E);
    gemm_wmma_kernel<half_t, true><<<dim3(E / 128, M1 / 128, 1), 256, 0, stream>>>(
        X1n, Wqt, bq, Qh, M1, E, D1, 0, 0, 0, scale);
    gemm_wmma_kernel<half_t, true><<<dim3(E / 128, M2 / 128, 1), 256, 0, stream>>>(
        X2n, Wkt, bk, Kh, M2, E, D2, 0, 0, 0, 1.0f);
    gemm_wmma_kernel<half_t, true><<<dim3(E / 128, M2 / 128, 1), 256, 0, stream>>>(
        X2n, Wvt, bv, Vh, M2, E, D2, 0, 0, 0, 1.0f);

    // 4) V -> V^T per batch (so B fragments of P@V are contiguous)
    transpose_f16_kernel<<<dim3(E / 32, L2 / 32, B), tb, 0, stream>>>(Vh, Vt, L2, E);

    // 5) scores = Q @ K^T  (Bt == K row-major already)
    gemm_wmma_kernel<half_t, false><<<dim3(L2 / 128, L1 / 128, B), 256, 0, stream>>>(
        Qh, Kh, nullptr, Sh, L1, L2, E,
        (long long)L1 * E, (long long)L2 * E, (long long)L1 * L2, 1.0f);

    // 6) row softmax in place
    softmax_rows_kernel<<<B * L1, 256, 0, stream>>>(Sh, L2);

    // 7) ctx = P @ V  (Bt == V^T [E, L2])
    gemm_wmma_kernel<half_t, false><<<dim3(E / 128, L1 / 128, B), 256, 0, stream>>>(
        Sh, Vt, nullptr, Ctx, L1, E, L2,
        (long long)L1 * L2, (long long)E * L2, (long long)L1 * E, 1.0f);

    // 8) out = ctx @ Wo + bo  (fp32 out)
    gemm_wmma_kernel<float, true><<<dim3(D1 / 128, M1 / 128, 1), 256, 0, stream>>>(
        Ctx, Wot, bo, out, M1, D1, E, 0, 0, 0, 1.0f);
}
